// F_ALSHConv2d_11390253269181
// MI455X (gfx1250) — matrix operand order repaired and checked by reading, not compiled, not run on hardware
//
#include <hip/hip_runtime.h>
#include <hip/hip_bf16.h>

// ---------------- static problem constants ----------------
#define BATCH   32
#define CIN     128
#define HDIM    56
#define WDIM    56
#define OCH     256
#define KSZ     3
#define NTAP    9            // KSZ*KSZ
#define KTOT    1152         // CIN*NTAP
#define MALS    9
#define ADIM    1161         // KTOT + MALS
#define NHASH   5
#define TSIZE   2048
#define RW      4.0f
#define UCONST  0.99f
#define NPIX    (BATCH*HDIM*WDIM)   // 100352
#define HW      (HDIM*WDIM)         // 3136
#define NCHUNK  (KTOT/32)           // 36

// ---------------- workspace layout (bytes) ----------------
#define WS_WHG      0                       // f16 [36][256][32]  = 589824 B
#define WS_COUNTS   589824                  // int [5][2048]      =  40960 B
#define WS_SEL      630784                  // int [8]
#define WS_KBUCK    630816                  // int [256][5]       =   5120 B
#define WS_CSCALE   635936                  // float [256]

typedef __attribute__((ext_vector_type(16))) _Float16 v16h;
typedef __attribute__((ext_vector_type(8)))  _Float16 v8h;
typedef __attribute__((ext_vector_type(8)))  float    v8f;

// CDNA5 async global->LDS copy of 16B per lane (ASYNCcnt-tracked, no VGPR data path)
__device__ __forceinline__ void async_copy_b128(const _Float16* gsrc, _Float16* ldst) {
    unsigned lds_off = (unsigned)(unsigned long long)(const void*)ldst; // flat->LDS: addr[31:0]
    unsigned long long ga = (unsigned long long)gsrc;
    asm volatile("global_load_async_to_lds_b128 %0, %1, off"
                 :: "v"(lds_off), "v"(ga) : "memory");
}
__device__ __forceinline__ void wait_async0() {
    asm volatile("s_wait_asynccnt 0x0" ::: "memory");
}

// =========================================================
// 1) Convert weights f32 OIHW -> f16 packed [chunk][o][32]
//    with k = tap*128 + c
// =========================================================
__global__ void wcvt_kernel(const float* __restrict__ w, _Float16* __restrict__ whg) {
    int idx = blockIdx.x * 256 + threadIdx.x;          // 36*256*32 = 294912
    int ck  = idx >> 13;                               // /8192
    int rem = idx & 8191;
    int o   = rem >> 5;
    int kl  = rem & 31;
    int k   = ck * 32 + kl;
    int t   = k >> 7;                                  // tap
    int c   = k & 127;                                 // channel
    whg[idx] = (_Float16)w[o * KTOT + c * NTAP + t];
}

// =========================================================
// 2) Hash-table build: kernel_bucket[o][h]
// =========================================================
__global__ void prep_kernel(const float* __restrict__ w,
                            const float* __restrict__ a,
                            const float* __restrict__ b_unit,
                            int* __restrict__ kbuck) {
    __shared__ float red[256];
    int o = threadIdx.x;

    float s2 = 0.0f;
    const float* wr = w + o * KTOT;
    for (int i = 0; i < KTOT; ++i) s2 += wr[i] * wr[i];
    float mynorm = sqrtf(s2);
    red[o] = mynorm;
    __syncthreads();
    for (int s = 128; s > 0; s >>= 1) {
        if (o < s) red[o] = fmaxf(red[o], red[o + s]);
        __syncthreads();
    }
    float denom = red[0];
    float scale = UCONST / denom;
    float nrm   = mynorm * scale;                      // ||wU_o||

    for (int h = 0; h < NHASH; ++h) {
        const float* ah = a + h * ADIM;
        float dot = 0.0f;
        for (int i = 0; i < KTOT; ++i) dot += wr[i] * ah[i];
        dot *= scale;
        float p = nrm;
        #pragma unroll
        for (int j = 0; j < MALS; ++j) {               // nrm^(2^(j+1))
            p = p * p;
            dot += p * ah[KTOT + j];
        }
        float bh = b_unit[h] * RW;
        int hv = (int)floorf((dot + bh) * (1.0f / RW));
        int bk = hv % TSIZE; if (bk < 0) bk = -bk;
        kbuck[o * NHASH + h] = bk;
    }
}

// =========================================================
// 3) Vote: per-pixel 5-hash conv -> histogram (global atomics)
// =========================================================
__global__ void vote_kernel(const float* __restrict__ x,
                            const float* __restrict__ a,
                            const float* __restrict__ b_unit,
                            int* __restrict__ counts) {
    __shared__ float a_s[NHASH * ADIM];                // 23220 B
    for (int i = threadIdx.x; i < NHASH * ADIM; i += 256) a_s[i] = a[i];
    __syncthreads();

    int p = blockIdx.x * 256 + threadIdx.x;            // NPIX = 392*256
    int b  = p / HW;
    int r  = p - b * HW;
    int y  = r / WDIM;
    int xc = r - y * WDIM;

    float acc[NHASH] = {0.f, 0.f, 0.f, 0.f, 0.f};
    #pragma unroll
    for (int t = 0; t < NTAP; ++t) {
        int dy = t / 3 - 1, dx = t % 3 - 1;
        int yy = y + dy, xx = xc + dx;
        if (yy < 0 || yy >= HDIM || xx < 0 || xx >= WDIM) continue;
        const float* xb = x + ((long)b * CIN * HW) + yy * WDIM + xx;
        for (int c = 0; c < CIN; ++c) {
            float xv = xb[c * HW];
            #pragma unroll
            for (int h = 0; h < NHASH; ++h)
                acc[h] += xv * a_s[h * ADIM + c * NTAP + t];
        }
        #pragma unroll
        for (int h = 0; h < NHASH; ++h)                // appended 0.5 channel
            acc[h] += 0.5f * a_s[h * ADIM + CIN * NTAP + t];
    }
    #pragma unroll
    for (int h = 0; h < NHASH; ++h) {
        float bh = b_unit[h] * RW;
        int hv = (int)floorf((acc[h] + bh) * (1.0f / RW));
        int bk = hv % TSIZE; if (bk < 0) bk = -bk;
        atomicAdd(&counts[h * TSIZE + bk], 1);
    }
}

// =========================================================
// 4) Per-hash argmax over histogram (first max index)
// =========================================================
__global__ void argmax_kernel(const int* __restrict__ counts, int* __restrict__ sel) {
    __shared__ int bc[256], bi[256];
    int tid = threadIdx.x;
    for (int h = 0; h < NHASH; ++h) {
        int mc = -1, mi = 0;
        for (int i = tid; i < TSIZE; i += 256) {
            int c = counts[h * TSIZE + i];
            if (c > mc || (c == mc && i < mi)) { mc = c; mi = i; }
        }
        bc[tid] = mc; bi[tid] = mi;
        __syncthreads();
        for (int s = 128; s > 0; s >>= 1) {
            if (tid < s) {
                int c2 = bc[tid + s], i2 = bi[tid + s];
                if (c2 > bc[tid] || (c2 == bc[tid] && i2 < bi[tid])) { bc[tid] = c2; bi[tid] = i2; }
            }
            __syncthreads();
        }
        if (tid == 0) sel[h] = bi[0];
        __syncthreads();
    }
}

// =========================================================
// 5) Active mask + per-channel scale (with empty fallback)
// =========================================================
__global__ void active_kernel(const int* __restrict__ kbuck,
                              const int* __restrict__ sel,
                              float* __restrict__ cscale) {
    __shared__ int ssel[NHASH];
    __shared__ int cnt[256];
    int o = threadIdx.x;
    if (o < NHASH) ssel[o] = sel[o];
    __syncthreads();

    bool act = false;
    #pragma unroll
    for (int h = 0; h < NHASH; ++h) {
        int kb = kbuck[o * NHASH + h];
        #pragma unroll
        for (int h2 = 0; h2 < NHASH; ++h2) act = act || (kb == ssel[h2]);
    }
    cnt[o] = act ? 1 : 0;
    __syncthreads();
    for (int s = 128; s > 0; s >>= 1) {
        if (o < s) cnt[o] += cnt[o + s];
        __syncthreads();
    }
    int n = cnt[0];
    if (n == 0) { act = true; n = OCH; }               // fallback: all active
    cscale[o] = act ? ((float)OCH / (float)n) : 0.0f;
}

// =========================================================
// 6) Main conv: implicit GEMM, f16 WMMA, f32 accumulate
//    A = weights [256 x K] (async global->LDS), B = im2col(x)
//    block: 256 o x 128 px, 8 waves; each wave 2x(16 o) x 128 px
//    -> every B fragment feeds 2 WMMAs; x read exactly once
//    double-buffered LDS; prefetch chunk ck+1 under WMMA(ck)
// =========================================================
__global__ void __launch_bounds__(256)
conv_main_kernel(const float* __restrict__ x,
                 const _Float16* __restrict__ whg,
                 const float* __restrict__ bias,
                 const float* __restrict__ cscale,
                 float* __restrict__ out) {
    __shared__ _Float16 As[2][256][40];                // weight tiles  (40960 B)
    __shared__ _Float16 Bs[2][128][40];                // patch tiles   (20480 B)

    const int tid     = threadIdx.x;
    const int lane    = tid & 31;
    const int wave    = tid >> 5;                      // 0..7 -> o sub-tile
    const int pixBase = blockIdx.x * 128;
    const int nlo     = lane & 15;                     // D-tile column / A row-in-tile
    const int hi      = lane >> 4;

    // ---- loop-invariant im2col coordinates (pl = tid&127 is chunk-invariant) ----
    const int pl  = tid & 127;
    const int klb = tid >> 7;                          // 0/1 -> K segment [klb*16, klb*16+16)
    const int p0  = pixBase + pl;
    const int bI  = p0 / HW;
    const int rpx = p0 - bI * HW;
    const int py  = rpx / WDIM;
    const int px  = rpx - py * WDIM;
    const float* xpix = x + (long)bI * CIN * HW;

    v8f acc[16];
    #pragma unroll
    for (int j = 0; j < 16; ++j) acc[j] = (v8f)(0.0f);

    auto stage = [&](int ck, int pb) {
        const int k0 = ck * 32;
        const int t  = k0 >> 7;
        const int dy = t / 3 - 1, dx = t % 3 - 1;
        const int c0 = k0 & 127;
        // A tile (256x32 f16 = 16KB): fire-and-forget async copies (ASYNCcnt)
        const _Float16* asrc = whg + (long)ck * (OCH * 32);
        #pragma unroll
        for (int i = 0; i < 4; ++i) {
            int cidx = tid + i * 256;                  // 1024 chunks of 8 halves
            int ol   = cidx >> 2;
            int kseg = (cidx & 3) * 8;
            async_copy_b128(asrc + ol * 32 + kseg, &As[pb][ol][kseg]);
        }
        // B tile: im2col + f32->f16, packed into 2x ds_store_b128 per thread
        const int yy = py + dy, xx = px + dx;
        const bool valid = (yy >= 0) && (yy < HDIM) && (xx >= 0) && (xx < WDIM);
        const int rowoff = yy * WDIM + xx;
        const float* src = xpix + (long)(c0 + klb * 16) * HW + rowoff;
        v8h pk0, pk1;
        #pragma unroll
        for (int i = 0; i < 8; ++i)
            pk0[i] = (_Float16)(valid ? src[(long)i * HW] : 0.0f);
        #pragma unroll
        for (int i = 0; i < 8; ++i)
            pk1[i] = (_Float16)(valid ? src[(long)(8 + i) * HW] : 0.0f);
        *reinterpret_cast<v8h*>(&Bs[pb][pl][klb * 16])     = pk0;
        *reinterpret_cast<v8h*>(&Bs[pb][pl][klb * 16 + 8]) = pk1;
    };

    stage(0, 0);
    int pb = 0;
    for (int ck = 0; ck < NCHUNK; ++ck) {
        wait_async0();                                 // A-tile async copies landed
        __syncthreads();                               // stage(pb) visible to all waves

        // ---- A fragments: rows m and m+128 ----
        const int m   = wave * 16 + nlo;
        const int k0a = hi * 8;                        // lane<16 K{0-7,16-23}, hi K{8-15,24-31}
        v8h a0l = *reinterpret_cast<const v8h*>(&As[pb][m][k0a]);
        v8h a0h = *reinterpret_cast<const v8h*>(&As[pb][m][k0a + 16]);
        v16h af0 = __builtin_shufflevector(a0l, a0h, 0,1,2,3,4,5,6,7,8,9,10,11,12,13,14,15);
        v8h a1l = *reinterpret_cast<const v8h*>(&As[pb][128 + m][k0a]);
        v8h a1h = *reinterpret_cast<const v8h*>(&As[pb][128 + m][k0a + 16]);
        v16h af1 = __builtin_shufflevector(a1l, a1h, 0,1,2,3,4,5,6,7,8,9,10,11,12,13,14,15);

        // ---- prefetch next chunk into buf[pb^1] while WMMAs run ----
        if (ck + 1 < NCHUNK) stage(ck + 1, pb ^ 1);

        // ---- each B fragment feeds two matrix ops ----
        const int kr = hi * 8;                         // lane<16 K{0-7,16-23}, hi K{8-15,24-31}
        #pragma unroll
        for (int j = 0; j < 8; ++j) {
            const int n = j * 16 + nlo;
            v8h blo = *reinterpret_cast<const v8h*>(&Bs[pb][n][kr]);
            v8h bhi = *reinterpret_cast<const v8h*>(&Bs[pb][n][kr + 16]);
            v16h bf = __builtin_shufflevector(blo, bhi, 0,1,2,3,4,5,6,7,8,9,10,11,12,13,14,15);
            acc[j]     = __builtin_amdgcn_wmma_f32_16x16x32_f16(
                false, af0, false, bf, (short)0, acc[j], false, false);
            acc[8 + j] = __builtin_amdgcn_wmma_f32_16x16x32_f16(
                false, af1, false, bf, (short)0, acc[8 + j], false, false);
        }
        pb ^= 1;
    }

    // ---- epilogue: (acc + bias) * channel_scale, coalesced on pixels ----
    float bi16[16], sc16[16];
    #pragma unroll
    for (int part = 0; part < 2; ++part)
        #pragma unroll
        for (int r = 0; r < 8; ++r) {
            int o = part * 128 + wave * 16 + r + 8 * hi; // D layout: VGPR r -> M = r + 8*hi
            bi16[part * 8 + r] = bias[o];
            sc16[part * 8 + r] = cscale[o];
        }
    #pragma unroll
    for (int j = 0; j < 8; ++j) {
        int p  = pixBase + j * 16 + nlo;               // N = lane&15
        int b2 = p / HW;
        int r2 = p - b2 * HW;
        long ob = (long)b2 * OCH;
        #pragma unroll
        for (int part = 0; part < 2; ++part)
            #pragma unroll
            for (int r = 0; r < 8; ++r) {
                int o = part * 128 + wave * 16 + r + 8 * hi;
                out[(ob + o) * HW + r2] =
                    (acc[part * 8 + j][r] + bi16[part * 8 + r]) * sc16[part * 8 + r];
            }
    }
}

// =========================================================
extern "C" void kernel_launch(void* const* d_in, const int* in_sizes, int n_in,
                              void* d_out, int out_size, void* d_ws, size_t ws_size,
                              hipStream_t stream) {
    const float* x      = (const float*)d_in[0];
    const float* weight = (const float*)d_in[1];
    const float* bias   = (const float*)d_in[2];
    const float* a      = (const float*)d_in[3];
    const float* b_unit = (const float*)d_in[4];
    float* out = (float*)d_out;

    char* ws = (char*)d_ws;
    _Float16* whg  = (_Float16*)(ws + WS_WHG);
    int* counts    = (int*)(ws + WS_COUNTS);
    int* sel       = (int*)(ws + WS_SEL);
    int* kbuck     = (int*)(ws + WS_KBUCK);
    float* cscale  = (float*)(ws + WS_CSCALE);

    // 1) pack weights to f16 [chunk][o][32]
    wcvt_kernel<<<(NCHUNK * OCH * 32) / 256, 256, 0, stream>>>(weight, whg);
    // 2) table build
    prep_kernel<<<1, 256, 0, stream>>>(weight, a, b_unit, kbuck);
    // 3) zero histogram, then vote
    hipMemsetAsync(counts, 0, NHASH * TSIZE * sizeof(int), stream);
    vote_kernel<<<NPIX / 256, 256, 0, stream>>>(x, a, b_unit, counts);
    // 4) per-hash argmax
    argmax_kernel<<<1, 256, 0, stream>>>(counts, sel);
    // 5) active mask + scale
    active_kernel<<<1, 256, 0, stream>>>(kbuck, sel, cscale);
    // 6) main conv via WMMA implicit GEMM (256-o tile, double-buffered, async A-stage)
    conv_main_kernel<<<NPIX / 128, 256, 0, stream>>>(x, whg, bias, cscale, out);
}
